// AdaLNCrossAttn_51702816309661
// MI455X (gfx1250) — matrix operand order repaired and checked by reading, not compiled, not run on hardware
//
#include <hip/hip_runtime.h>

// MI455X (gfx1250) wave32 WMMA implementation of the AdaLN cross-attention block.
// GEMM contractions via v_wmma_f32_16x16x32_bf16; LDS tile fills via the CDNA5
// async copy path (global_load_async_to_lds_b128 + s_wait_asynccnt), with a
// double-buffered GEMM main loop.

typedef __attribute__((ext_vector_type(16))) __bf16        v16bf;
typedef __attribute__((ext_vector_type(8)))  float         v8f;
typedef __attribute__((ext_vector_type(4)))  unsigned int  u32x4;

union Frag16 { v16bf v; u32x4 u[2]; };

__device__ __forceinline__ v8f wmma_bf16(v16bf a, v16bf b, v8f c) {
  return __builtin_amdgcn_wmma_f32_16x16x32_bf16(false, a, false, b, (short)0, c,
                                                 false, false);
}

// Generic pointers to LDS are {aperture_hi, lds_offset32}; truncation yields the
// LDS byte address expected by async-to-LDS VDST.
__device__ __forceinline__ unsigned lds_off(const void* p) {
  return (unsigned)(size_t)p;
}

// CDNA5 async copy: 16B per lane, global -> LDS, tracked by ASYNCcnt.
__device__ __forceinline__ void async_load_b128(unsigned lds, const void* gptr) {
  asm volatile("global_load_async_to_lds_b128 %0, %1, off"
               :
               : "v"(lds), "v"((unsigned long long)(size_t)gptr)
               : "memory");
}

__device__ __forceinline__ void wait_async0() {
  asm volatile("s_wait_asynccnt 0x0" ::: "memory");
}

// ---------------------------------------------------------------------------
// fp32 -> bf16 conversion (weights / context), grid-stride
// ---------------------------------------------------------------------------
__global__ __launch_bounds__(256) void f32_to_bf16(const float* __restrict__ in,
                                                   __bf16* __restrict__ out,
                                                   long n) {
  long i = (long)blockIdx.x * blockDim.x + threadIdx.x;
  long stride = (long)gridDim.x * blockDim.x;
  for (; i < n; i += stride) out[i] = (__bf16)in[i];
}

// ---------------------------------------------------------------------------
// Fused LayerNorm + AdaLN modulation: out = LN(x)*(1+scale) + shift  (bf16)
// ---------------------------------------------------------------------------
__global__ __launch_bounds__(256) void ln_mod(const float* __restrict__ x,
                                              const float* __restrict__ condBD,
                                              const float* __restrict__ adaGss,
                                              int scaleIdx, int shiftIdx,
                                              __bf16* __restrict__ out) {
  constexpr int Cc = 1024, Lr = 1024;
  const int row = blockIdx.x;
  const int b = row / Lr;
  const float* xr = x + (size_t)row * Cc;

  float s = 0.f, s2 = 0.f;
  for (int c = threadIdx.x; c < Cc; c += 256) {
    float v = xr[c];
    s += v; s2 += v * v;
  }
#pragma unroll
  for (int m = 16; m >= 1; m >>= 1) {
    s  += __shfl_xor(s, m, 32);
    s2 += __shfl_xor(s2, m, 32);
  }
  __shared__ float shs[8], sh2[8];
  const int wave = threadIdx.x >> 5, lane = threadIdx.x & 31;
  if (lane == 0) { shs[wave] = s; sh2[wave] = s2; }
  __syncthreads();
  float ts = 0.f, ts2 = 0.f;
#pragma unroll
  for (int i = 0; i < 8; ++i) { ts += shs[i]; ts2 += sh2[i]; }
  const float mean = ts * (1.f / Cc);
  const float var  = ts2 * (1.f / Cc) - mean * mean;
  const float rstd = rsqrtf(var + 1e-5f);

  const float* scA = adaGss + (size_t)scaleIdx * Cc;
  const float* shA = adaGss + (size_t)shiftIdx * Cc;
  const float* scC = condBD + ((size_t)b * 6 + scaleIdx) * Cc;
  const float* shC = condBD + ((size_t)b * 6 + shiftIdx) * Cc;
  for (int c = threadIdx.x; c < Cc; c += 256) {
    float sv = 1.f + scA[c] + scC[c];
    float sh = shA[c] + shC[c];
    out[(size_t)row * Cc + c] = (__bf16)(((xr[c] - mean) * rstd) * sv + sh);
  }
}

// ---------------------------------------------------------------------------
// NT WMMA GEMM: C[M,N] = A[M,K] @ B[N,K]^T, A/B bf16 row-major (K-major).
// 128x128x64 tile, 8 waves (2x4), per-wave 64x32. Double-buffered LDS fills
// via global_load_async_to_lds_b128.
// EPI: 0 = store bf16; 1 = gelu(acc+bias) bf16; 2 = resid + (acc+bias)*gamma f32
// ---------------------------------------------------------------------------
template <int EPI>
__global__ __launch_bounds__(256) void gemm_bf16_nt(
    const __bf16* __restrict__ A, const __bf16* __restrict__ Bw,
    int M, int N, int K,
    float* __restrict__ outF, __bf16* __restrict__ outH,
    const float* __restrict__ bias, const float* __restrict__ resid,
    const float* __restrict__ condBD, const float* __restrict__ adaGss,
    int gIdx, int Lrows) {
  (void)M;
  __shared__ __bf16 As[2][128][72];  // 64 cols + 8 pad
  __shared__ __bf16 Bs[2][128][72];

  const int tid  = threadIdx.x;
  const int wave = tid >> 5, lane = tid & 31;
  const int lm = lane & 15, hi = lane >> 4;
  const int wm = wave & 1, wn = wave >> 1;
  const int m0 = blockIdx.y * 128, n0 = blockIdx.x * 128;

  v8f acc[4][2] = {};

  const int ldr = tid >> 2;         // 0..63
  const int ldc = (tid & 3) * 16;   // 0,16,32,48

  auto issue_tile = [&](int buf, int k0) {
#pragma unroll
    for (int rr = 0; rr < 2; ++rr) {
      const int row = ldr + rr * 64;
      async_load_b128(lds_off(&As[buf][row][ldc]),
                      A + (size_t)(m0 + row) * K + k0 + ldc);
      async_load_b128(lds_off(&Bs[buf][row][ldc]),
                      Bw + (size_t)(n0 + row) * K + k0 + ldc);
    }
  };

  const int nk = K / 64;
  issue_tile(0, 0);

  for (int kt = 0; kt < nk; ++kt) {
    wait_async0();        // this wave's fills for tile kt have landed
    __syncthreads();      // => everyone's fills landed; prev buffer free
    if (kt + 1 < nk) issue_tile((kt + 1) & 1, (kt + 1) * 64);
    const int buf = kt & 1;
#pragma unroll
    for (int ks = 0; ks < 2; ++ks) {
      Frag16 af[4], bf[2];
#pragma unroll
      for (int mi = 0; mi < 4; ++mi) {
        const int row = wm * 64 + mi * 16 + lm;
        af[mi].u[0] = *(const u32x4*)&As[buf][row][ks * 32 + hi * 8];
        af[mi].u[1] = *(const u32x4*)&As[buf][row][ks * 32 + 16 + hi * 8];
      }
#pragma unroll
      for (int nj = 0; nj < 2; ++nj) {
        const int nrow = wn * 32 + nj * 16 + lm;
        bf[nj].u[0] = *(const u32x4*)&Bs[buf][nrow][ks * 32 + hi * 8];
        bf[nj].u[1] = *(const u32x4*)&Bs[buf][nrow][ks * 32 + 16 + hi * 8];
      }
#pragma unroll
      for (int mi = 0; mi < 4; ++mi)
#pragma unroll
        for (int nj = 0; nj < 2; ++nj)
          acc[mi][nj] = wmma_bf16(af[mi].v, bf[nj].v, acc[mi][nj]);
    }
    __syncthreads();      // all waves done reading buf before it is refilled
  }

  // Epilogue. C-frag: lanes 0-15 row r, lanes 16-31 row r+8; col = lane%16.
#pragma unroll
  for (int nj = 0; nj < 2; ++nj) {
    const int col = n0 + wn * 32 + nj * 16 + lm;
    float bv = 0.f;
    if constexpr (EPI == 1 || EPI == 2) bv = bias[col];
#pragma unroll
    for (int mi = 0; mi < 4; ++mi) {
#pragma unroll
      for (int r = 0; r < 8; ++r) {
        const int row = m0 + wm * 64 + mi * 16 + hi * 8 + r;
        float v = acc[mi][nj][r];
        if constexpr (EPI == 0) {
          outH[(size_t)row * N + col] = (__bf16)v;
        } else if constexpr (EPI == 1) {
          v += bv;
          float g = 0.5f * v *
                    (1.f + tanhf(0.7978845608028654f * (v + 0.044715f * v * v * v)));
          outH[(size_t)row * N + col] = (__bf16)g;
        } else {
          const int bb = row / Lrows;
          const float gam = adaGss[(size_t)gIdx * N + col] +
                            condBD[((size_t)bb * 6 + gIdx) * N + col];
          outF[(size_t)row * N + col] =
              resid[(size_t)row * N + col] + (v + bv) * gam;
        }
      }
    }
  }
}

// ---------------------------------------------------------------------------
// Flash-style cross attention with bias: softmax(Q K^T * SCALE + bias) V.
// Block = one (b,h) x 128 Q-rows; each wave owns 16 rows. LC streamed in
// 32-wide tiles (K async-copied to LDS, V transposed in LDS). Online softmax.
// ---------------------------------------------------------------------------
__global__ __launch_bounds__(256) void attn_fwd(const __bf16* __restrict__ Q,
                                                const __bf16* __restrict__ Kg,
                                                const __bf16* __restrict__ Vg,
                                                const float* __restrict__ bias,
                                                __bf16* __restrict__ O) {
  constexpr int L = 1024, LC = 1024, H = 16, HD = 64, C = 1024;
  constexpr float SCALE = 0.03125f;  // 0.25 / sqrt(64)
  const int b = blockIdx.z, h = blockIdx.y;
  const int wave = threadIdx.x >> 5, lane = threadIdx.x & 31;
  const int lm = lane & 15, hi = lane >> 4;
  const int m0 = blockIdx.x * 128 + wave * 16;

  __shared__ __bf16 Ks[32][72];      // [lc][hd]
  __shared__ __bf16 Vt[64][40];      // [hd][lc]  (transposed)
  __shared__ __bf16 Ps[8][16][40];   // per-wave P staging (C-layout -> A-layout)

  Frag16 qa[2];
  {
    const __bf16* qrow = Q + ((size_t)b * L + m0 + lm) * C + h * HD;
#pragma unroll
    for (int ks = 0; ks < 2; ++ks) {
      qa[ks].u[0] = *(const u32x4*)(qrow + ks * 32 + hi * 8);
      qa[ks].u[1] = *(const u32x4*)(qrow + ks * 32 + 16 + hi * 8);
    }
  }

  v8f accO[4] = {};
  float mrow[8], lrow[8];
#pragma unroll
  for (int r = 0; r < 8; ++r) { mrow[r] = -1e30f; lrow[r] = 0.f; }

  const int ldr = threadIdx.x >> 3;        // 0..31  (lc row within tile)
  const int ldc = (threadIdx.x & 7) * 8;   // 0..56  (hd col)

  for (int t = 0; t < LC; t += 32) {
    __syncthreads();   // everyone done reading previous Ks/Vt
    {
      const size_t base = ((size_t)b * LC + t + ldr) * C + h * HD + ldc;
      async_load_b128(lds_off(&Ks[ldr][ldc]), Kg + base);
      union { u32x4 u; __bf16 e[8]; } vl;
      vl.u = *(const u32x4*)(Vg + base);
#pragma unroll
      for (int j = 0; j < 8; ++j) Vt[ldc + j][ldr] = vl.e[j];
    }
    wait_async0();
    __syncthreads();

    // S = Q K^T  (16 x 32 per wave, K-dim = HD = 64 -> 2 wmma k-steps)
    v8f sAcc[2] = {};
#pragma unroll
    for (int ks = 0; ks < 2; ++ks) {
#pragma unroll
      for (int nj = 0; nj < 2; ++nj) {
        Frag16 kf;
        const int n = nj * 16 + lm;
        kf.u[0] = *(const u32x4*)&Ks[n][ks * 32 + hi * 8];
        kf.u[1] = *(const u32x4*)&Ks[n][ks * 32 + 16 + hi * 8];
        sAcc[nj] = wmma_bf16(qa[ks].v, kf.v, sAcc[nj]);
      }
    }

    // bias + scale + online softmax (row r+hi*8 lives in this lane half)
    const float* bp = bias + (((size_t)b * H + h) * L + m0 + hi * 8) * LC + t;
    float scl[8];
#pragma unroll
    for (int r = 0; r < 8; ++r) {
      float s0 = sAcc[0][r] * SCALE + bp[(size_t)r * LC + lm];
      float s1 = sAcc[1][r] * SCALE + bp[(size_t)r * LC + 16 + lm];
      float mx = fmaxf(s0, s1);
      mx = fmaxf(mx, __shfl_xor(mx, 1, 32));
      mx = fmaxf(mx, __shfl_xor(mx, 2, 32));
      mx = fmaxf(mx, __shfl_xor(mx, 4, 32));
      mx = fmaxf(mx, __shfl_xor(mx, 8, 32));
      const float nm = fmaxf(mrow[r], mx);
      const float sc = __expf(mrow[r] - nm);
      const float p0 = __expf(s0 - nm);
      const float p1 = __expf(s1 - nm);
      float rs = p0 + p1;
      rs += __shfl_xor(rs, 1, 32);
      rs += __shfl_xor(rs, 2, 32);
      rs += __shfl_xor(rs, 4, 32);
      rs += __shfl_xor(rs, 8, 32);
      lrow[r] = lrow[r] * sc + rs;
      mrow[r] = nm;
      scl[r] = sc;
      Ps[wave][hi * 8 + r][lm]      = (__bf16)p0;
      Ps[wave][hi * 8 + r][16 + lm] = (__bf16)p1;
    }
#pragma unroll
    for (int dj = 0; dj < 4; ++dj)
#pragma unroll
      for (int r = 0; r < 8; ++r) accO[dj][r] *= scl[r];

    // O += P V  (K-dim = 32 lc, N = 64 hd -> 4 n-blocks)
    Frag16 pa;
    pa.u[0] = *(const u32x4*)&Ps[wave][lm][hi * 8];
    pa.u[1] = *(const u32x4*)&Ps[wave][lm][16 + hi * 8];
#pragma unroll
    for (int dj = 0; dj < 4; ++dj) {
      Frag16 vf;
      const int n = dj * 16 + lm;
      vf.u[0] = *(const u32x4*)&Vt[n][hi * 8];
      vf.u[1] = *(const u32x4*)&Vt[n][16 + hi * 8];
      accO[dj] = wmma_bf16(pa.v, vf.v, accO[dj]);
    }
  }

#pragma unroll
  for (int dj = 0; dj < 4; ++dj) {
#pragma unroll
    for (int r = 0; r < 8; ++r) {
      const float inv = 1.f / lrow[r];
      const int row = m0 + hi * 8 + r;
      O[((size_t)b * L + row) * C + h * HD + dj * 16 + lm] =
          (__bf16)(accO[dj][r] * inv);
    }
  }
}

// ---------------------------------------------------------------------------
// Host launcher
// ---------------------------------------------------------------------------
extern "C" void kernel_launch(void* const* d_in, const int* in_sizes, int n_in,
                              void* d_out, int out_size, void* d_ws,
                              size_t ws_size, hipStream_t stream) {
  (void)in_sizes; (void)n_in; (void)out_size; (void)ws_size;
  constexpr int B = 4, L = 1024, LC = 1024, C = 1024, H = 16, F = 4096;
  constexpr int M = B * L;

  const float* x      = (const float*)d_in[0];
  const float* ctx    = (const float*)d_in[1];
  const float* condBD = (const float*)d_in[2];
  const float* abias  = (const float*)d_in[3];
  const float* adaGss = (const float*)d_in[4];
  const float* Wq     = (const float*)d_in[5];
  const float* Wk     = (const float*)d_in[6];
  const float* Wv     = (const float*)d_in[7];
  const float* Wo     = (const float*)d_in[8];
  const float* bo     = (const float*)d_in[9];
  const float* W1     = (const float*)d_in[10];
  const float* b1     = (const float*)d_in[11];
  const float* W2     = (const float*)d_in[12];
  const float* b2     = (const float*)d_in[13];
  float* outp = (float*)d_out;

  char* w = (char*)d_ws;
  auto alloc = [&](size_t bytes) -> char* {
    char* p = w;
    w += (bytes + 255) & ~(size_t)255;
    return p;
  };
  __bf16* ctxb = (__bf16*)alloc((size_t)B * LC * C * 2);
  __bf16* modx = (__bf16*)alloc((size_t)M * C * 2);
  __bf16* wqb  = (__bf16*)alloc((size_t)C * C * 2);
  __bf16* wkb  = (__bf16*)alloc((size_t)C * C * 2);
  __bf16* wvb  = (__bf16*)alloc((size_t)C * C * 2);
  __bf16* wob  = (__bf16*)alloc((size_t)C * C * 2);
  __bf16* w1b  = (__bf16*)alloc((size_t)F * C * 2);
  __bf16* w2b  = (__bf16*)alloc((size_t)C * F * 2);
  __bf16* qb   = (__bf16*)alloc((size_t)M * C * 2);
  __bf16* kb   = (__bf16*)alloc((size_t)B * LC * C * 2);
  __bf16* vb   = (__bf16*)alloc((size_t)B * LC * C * 2);
  __bf16* ob   = (__bf16*)alloc((size_t)M * C * 2);
  float*  x1   = (float*)alloc((size_t)M * C * 4);
  __bf16* modf = (__bf16*)alloc((size_t)M * C * 2);
  __bf16* hb   = (__bf16*)alloc((size_t)M * F * 2);

  // 1) precision conversion of operands
  f32_to_bf16<<<2048, 256, 0, stream>>>(ctx, ctxb, (long)B * LC * C);
  f32_to_bf16<<<1024, 256, 0, stream>>>(Wq, wqb, (long)C * C);
  f32_to_bf16<<<1024, 256, 0, stream>>>(Wk, wkb, (long)C * C);
  f32_to_bf16<<<1024, 256, 0, stream>>>(Wv, wvb, (long)C * C);
  f32_to_bf16<<<1024, 256, 0, stream>>>(Wo, wob, (long)C * C);
  f32_to_bf16<<<2048, 256, 0, stream>>>(W1, w1b, (long)F * C);
  f32_to_bf16<<<2048, 256, 0, stream>>>(W2, w2b, (long)C * F);

  // 2) LN + AdaLN modulation (scale1=idx2, shift1=idx4)
  ln_mod<<<M, 256, 0, stream>>>(x, condBD, adaGss, 2, 4, modx);

  // 3) Q/K/V projections
  gemm_bf16_nt<0><<<dim3(C / 128, M / 128), 256, 0, stream>>>(
      modx, wqb, M, C, C, nullptr, qb, nullptr, nullptr, nullptr, nullptr, 0, L);
  gemm_bf16_nt<0><<<dim3(C / 128, (B * LC) / 128), 256, 0, stream>>>(
      ctxb, wkb, B * LC, C, C, nullptr, kb, nullptr, nullptr, nullptr, nullptr, 0, L);
  gemm_bf16_nt<0><<<dim3(C / 128, (B * LC) / 128), 256, 0, stream>>>(
      ctxb, wvb, B * LC, C, C, nullptr, vb, nullptr, nullptr, nullptr, nullptr, 0, L);

  // 4) attention with bias + softmax
  attn_fwd<<<dim3(L / 128, H, B), 256, 0, stream>>>(qb, kb, vb, abias, ob);

  // 5) output projection + residual: x1 = x + (O@Wo^T + bo) * gamma1
  gemm_bf16_nt<2><<<dim3(C / 128, M / 128), 256, 0, stream>>>(
      ob, wob, M, C, C, x1, nullptr, bo, x, condBD, adaGss, 0, L);

  // 6) LN + modulation 2 (scale2=idx3, shift2=idx5)
  ln_mod<<<M, 256, 0, stream>>>(x1, condBD, adaGss, 3, 5, modf);

  // 7) FFN up + GELU
  gemm_bf16_nt<1><<<dim3(F / 128, M / 128), 256, 0, stream>>>(
      modf, w1b, M, F, C, nullptr, hb, b1, nullptr, nullptr, nullptr, 0, L);

  // 8) FFN down + residual: out = x1 + (H@W2^T + b2) * gamma2
  gemm_bf16_nt<2><<<dim3(C / 128, M / 128), 256, 0, stream>>>(
      hb, w2b, M, C, F, outp, nullptr, b2, x1, condBD, adaGss, 1, L);
}
